// EdgeEncoder_4544075399549
// MI455X (gfx1250) — matrix operand order repaired and checked by reading, not compiled, not run on hardware
//
#include <hip/hip_runtime.h>
#include <cstdint>
#include <cstddef>

// ---------- types ----------
typedef __attribute__((ext_vector_type(16))) __bf16 v16bf;
typedef __attribute__((ext_vector_type(8)))  __bf16 v8bf;
typedef __attribute__((ext_vector_type(8)))  float  v8f;
typedef __attribute__((ext_vector_type(2)))  float  v2f;

#define Dm    256
#define Tt    9
#define WAVES 2
#define HPAD  264   // 256 + 8 bf16 pad -> 528B row stride, kills LDS bank conflicts

// SCALES from the reference (compile-time constants)
__constant__ float c_scales[Tt][4] = {
  {1.f,    1e-6f, 1.f,  1.f},  // nmos
  {1.f,    1e-6f, 1.f,  1.f},  // pmos
  {1.f,    1.f,   1.f,  1.f},  // balun
  {1e3f,   1.f,   1.f,  1.f},  // resistor
  {1e-12f, 1.f,   1.f,  1.f},  // capacitor
  {1e-9f,  1.f,   1.f,  1.f},  // inductor
  {1.f,    1.f,   1.f,  1.f},  // vsource
  {1e-3f,  1e-3f, 1.f,  1.f},  // isource
  {1.f,    1.f,   1e9f, 1.f},  // port
};

// ---------- sort-by-type prep ----------
__global__ void k_hist(const int* __restrict__ base_ids, int E, int* __restrict__ hist) {
  int i = blockIdx.x * blockDim.x + threadIdx.x;
  if (i < E) atomicAdd(&hist[base_ids[i]], 1);
}

__global__ void k_scan(const int* __restrict__ hist, int* __restrict__ cursor) {
  if (threadIdx.x == 0) {
    int s = 0;
    for (int t = 0; t < Tt; ++t) { cursor[t] = s; s += hist[t]; }
  }
}

__global__ void k_scatter(const int* __restrict__ base_ids, int E,
                          int* __restrict__ cursor, int* __restrict__ order) {
  int i = blockIdx.x * blockDim.x + threadIdx.x;
  if (i < E) {
    int p = atomicAdd(&cursor[base_ids[i]], 1);
    order[p] = i;
  }
}

// ---------- weight pre-layout: f32 row-major -> bf16 WMMA B-fragments ----------
// Fragment (kt,nt) covers K=[kt*32,kt*32+32), N=[nt*16,nt*16+16).
// Per-lane 16 contiguous bf16: lane l -> column n = nt*16 + (l&15),
// element j -> K = kt*32 + (l>>4)*8 + j for j<8, kt*32 + 16 + (l>>4)*8 + (j-8) for j>=8.
__global__ void k_wprep(const float* __restrict__ W2, const float* __restrict__ Wf,
                        __bf16* __restrict__ W2B, __bf16* __restrict__ WfB) {
  const int NW2 = Tt * 8 * 16 * 512;
  const int NWF = 8 * 16 * 512;
  int idx = blockIdx.x * blockDim.x + threadIdx.x;
  if (idx >= NW2 + NWF) return;
  int i2 = (idx < NW2) ? idx : idx - NW2;
  int j    =  i2        & 15;
  int lane = (i2 >> 4)  & 31;
  int nt   = (i2 >> 9)  & 15;
  int kt   = (i2 >> 13) & 7;
  int n = nt * 16 + (lane & 15);
  int k = kt * 32 + (lane >> 4) * 8 + (j & 7) + (j >> 3) * 16;
  if (idx < NW2) {
    int t = i2 >> 16;
    W2B[i2] = (__bf16)W2[(t * Dm + k) * Dm + n];
  } else {
    WfB[i2] = (__bf16)Wf[(512 + k) * Dm + n];   // bottom third of Wf
  }
}

// ---------- fold type/source embeddings through Wf (tiny: 18x256 outputs) ----------
__global__ void k_embed(const float* __restrict__ type_embed, const float* __restrict__ source_embed,
                        const float* __restrict__ Wf, const float* __restrict__ bf,
                        float* __restrict__ TEb, float* __restrict__ SEb) {
  int idx = blockIdx.x * blockDim.x + threadIdx.x;
  if (idx >= 18 * Dm) return;
  int row = idx >> 8, n = idx & 255;
  if (row < 14) {
    float s = bf[n];                          // fold bias into TEb
    for (int d = 0; d < Dm; ++d) s += type_embed[row * Dm + d] * Wf[d * Dm + n];
    TEb[row * Dm + n] = s;
  } else {
    int sr = row - 14;
    float s = 0.f;
    for (int d = 0; d < Dm; ++d) s += source_embed[sr * Dm + d] * Wf[(Dm + d) * Dm + n];
    SEb[sr * Dm + n] = s;
  }
}

// ---------- fused main kernel: one wave = one 16-edge (type-sorted) tile ----------
__global__ __launch_bounds__(WAVES * 32) void k_main(
    const int* __restrict__ type_ids, const int* __restrict__ source_ids,
    const int* __restrict__ base_ids, const float* __restrict__ params,
    const float* __restrict__ W1, const float* __restrict__ b1,
    const float* __restrict__ b2,
    const __bf16* __restrict__ W2B, const __bf16* __restrict__ WfB,
    const float* __restrict__ TEb, const float* __restrict__ SEb,
    const int* __restrict__ order, float* __restrict__ out, int E) {

  __shared__ __align__(16) __bf16 sh_h [WAVES][16][HPAD];
  __shared__ __align__(16) __bf16 sh_pv[WAVES][16][HPAD];
  __shared__ int4 sh_meta[WAVES][16];   // {edge, base, type, source}

  const int w    = threadIdx.x >> 5;
  const int l    = threadIdx.x & 31;
  const int m    = l & 15;
  const int hi   = l >> 4;
  const int tile = blockIdx.x * WAVES + w;

  if (l < 16) {
    int r = tile * 16 + l;
    int4 mt = make_int4(-1, 0, 0, 0);
    if (r < E) {
      int e = order[r];
      mt = make_int4(e, base_ids[e], type_ids[e], source_ids[e]);
    }
    sh_meta[w][l] = mt;
  }
  __syncthreads();

  int4 mm = sh_meta[w][m];              // this lane's row metadata
  const int tbm = mm.y;
  // valid-row mask via wave ballot (lanes 16..31 duplicate bits 0..15)
  unsigned rem0 = __builtin_amdgcn_ballot_w32(mm.x >= 0) & 0xffffu;

  // x as 16x4 f32 A-fragment: lane holds x[m][2*hi], x[m][2*hi+1]
  v2f xa; xa[0] = 0.f; xa[1] = 0.f;
  if (mm.x >= 0) {
    xa[0] = params[mm.x * 4 + 2 * hi + 0] / c_scales[tbm][2 * hi + 0];
    xa[1] = params[mm.x * 4 + 2 * hi + 1] / c_scales[tbm][2 * hi + 1];
  }

  // ---- Stage 1: h = relu(x @ W1[t] + b1[t])  via V_WMMA_F32_16X16X4_F32 ----
  {
    unsigned rem = rem0;
    while (rem) {
      int r0 = __builtin_ctz(rem);
      int t = sh_meta[w][r0].y;
      unsigned cur = __builtin_amdgcn_ballot_w32(((rem >> m) & 1) && (tbm == t)) & 0xffffu;
      rem &= ~cur;
      if (cur == 0xffffu) {             // uniform tile: unpredicated stores
        for (int nt = 0; nt < 16; ++nt) {
          int n = nt * 16 + m;
          v2f bfr;
          bfr[0] = W1[(t * 4 + 2 * hi + 0) * Dm + n];
          bfr[1] = W1[(t * 4 + 2 * hi + 1) * Dm + n];
          v8f hacc = {};
          hacc = __builtin_amdgcn_wmma_f32_16x16x4_f32(
              false, xa, false, bfr, (short)0, hacc, false, false);
          float bb = b1[t * Dm + n];
#pragma unroll
          for (int j = 0; j < 8; ++j)
            sh_h[w][j + 8 * hi][n] = (__bf16)fmaxf(hacc[j] + bb, 0.f);
        }
      } else {                          // boundary tile: write-predicated rows
        for (int nt = 0; nt < 16; ++nt) {
          int n = nt * 16 + m;
          v2f bfr;
          bfr[0] = W1[(t * 4 + 2 * hi + 0) * Dm + n];
          bfr[1] = W1[(t * 4 + 2 * hi + 1) * Dm + n];
          v8f hacc = {};
          hacc = __builtin_amdgcn_wmma_f32_16x16x4_f32(
              false, xa, false, bfr, (short)0, hacc, false, false);
          float bb = b1[t * Dm + n];
#pragma unroll
          for (int j = 0; j < 8; ++j) {
            int mj = j + 8 * hi;
            if ((cur >> mj) & 1)
              sh_h[w][mj][n] = (__bf16)fmaxf(hacc[j] + bb, 0.f);
          }
        }
      }
    }
  }
  __syncthreads();

  // reload h as 16x32 bf16 A-fragments (two 16B LDS reads per fragment per lane)
  v16bf ha[8];
#pragma unroll
  for (int kt = 0; kt < 8; ++kt) {
    int kb = kt * 32 + hi * 8;
    v8bf lo = *(const v8bf*)&sh_h[w][m][kb];
    v8bf hp = *(const v8bf*)&sh_h[w][m][kb + 16];
    ha[kt] = __builtin_shufflevector(lo, hp, 0,1,2,3,4,5,6,7,8,9,10,11,12,13,14,15);
  }

  // ---- Stage 2: pv = h @ W2[t] + b2[t]  via V_WMMA_F32_16X16X32_BF16 ----
  {
    unsigned rem = rem0;
    while (rem) {
      int r0 = __builtin_ctz(rem);
      int t = sh_meta[w][r0].y;
      unsigned cur = __builtin_amdgcn_ballot_w32(((rem >> m) & 1) && (tbm == t)) & 0xffffu;
      rem &= ~cur;
      const v16bf* wb = (const v16bf*)(W2B + (size_t)t * 8 * 16 * 512);
      if (cur == 0xffffu) {
        // fast path: 2 N-tiles in flight -> two independent WMMA chains
        for (int nt = 0; nt < 16; nt += 2) {
          v8f a0 = {}, a1 = {};
#pragma unroll
          for (int kt = 0; kt < 8; ++kt) {
            v16bf b0 = wb[(kt * 16 + nt + 0) * 32 + l];
            v16bf b1v = wb[(kt * 16 + nt + 1) * 32 + l];
            a0 = __builtin_amdgcn_wmma_f32_16x16x32_bf16(
                false, ha[kt], false, b0, (short)0, a0, false, false);
            a1 = __builtin_amdgcn_wmma_f32_16x16x32_bf16(
                false, ha[kt], false, b1v, (short)0, a1, false, false);
          }
          int n0 = nt * 16 + m, n1 = n0 + 16;
          float bb0 = b2[t * Dm + n0], bb1 = b2[t * Dm + n1];
#pragma unroll
          for (int j = 0; j < 8; ++j) {
            int mj = j + 8 * hi;
            sh_pv[w][mj][n0] = (__bf16)(a0[j] + bb0);
            sh_pv[w][mj][n1] = (__bf16)(a1[j] + bb1);
          }
        }
      } else {
        for (int nt = 0; nt < 16; ++nt) {
          v8f acc = {};
#pragma unroll
          for (int kt = 0; kt < 8; ++kt) {
            v16bf bmat = wb[(kt * 16 + nt) * 32 + l];
            acc = __builtin_amdgcn_wmma_f32_16x16x32_bf16(
                false, ha[kt], false, bmat, (short)0, acc, false, false);
          }
          int n = nt * 16 + m;
          float bb = b2[t * Dm + n];
#pragma unroll
          for (int j = 0; j < 8; ++j) {
            int mj = j + 8 * hi;
            if ((cur >> mj) & 1)
              sh_pv[w][mj][n] = (__bf16)(acc[j] + bb);
          }
        }
      }
    }
  }
  __syncthreads();

  // reload pv as A-fragments
  v16bf pa[8];
#pragma unroll
  for (int kt = 0; kt < 8; ++kt) {
    int kb = kt * 32 + hi * 8;
    v8bf lo = *(const v8bf*)&sh_pv[w][m][kb];
    v8bf hp = *(const v8bf*)&sh_pv[w][m][kb + 16];
    pa[kt] = __builtin_shufflevector(lo, hp, 0,1,2,3,4,5,6,7,8,9,10,11,12,13,14,15);
  }

  int rowE[8], rowTy[8], rowSr[8];
#pragma unroll
  for (int j = 0; j < 8; ++j) {
    int4 mr = sh_meta[w][j + 8 * hi];
    rowE[j] = mr.x; rowTy[j] = mr.z; rowSr[j] = mr.w;
  }

  // ---- Stage 3: out = relu(pv @ Wf[512:768] + TEb[type] + SEb[src]) ----
  const v16bf* fb = (const v16bf*)WfB;
  if (rem0 == 0xffffu) {
    // full tile: unpredicated stores, 2 N-tiles in flight
    for (int nt = 0; nt < 16; nt += 2) {
      v8f a0 = {}, a1 = {};
#pragma unroll
      for (int kt = 0; kt < 8; ++kt) {
        v16bf b0 = fb[(kt * 16 + nt + 0) * 32 + l];
        v16bf b1v = fb[(kt * 16 + nt + 1) * 32 + l];
        a0 = __builtin_amdgcn_wmma_f32_16x16x32_bf16(
            false, pa[kt], false, b0, (short)0, a0, false, false);
        a1 = __builtin_amdgcn_wmma_f32_16x16x32_bf16(
            false, pa[kt], false, b1v, (short)0, a1, false, false);
      }
      int n0 = nt * 16 + m, n1 = n0 + 16;
#pragma unroll
      for (int j = 0; j < 8; ++j) {
        float v0 = a0[j] + TEb[rowTy[j] * Dm + n0] + SEb[rowSr[j] * Dm + n0];
        float v1 = a1[j] + TEb[rowTy[j] * Dm + n1] + SEb[rowSr[j] * Dm + n1];
        size_t rb = (size_t)rowE[j] * Dm;
        out[rb + n0] = fmaxf(v0, 0.f);
        out[rb + n1] = fmaxf(v1, 0.f);
      }
    }
  } else {
    for (int nt = 0; nt < 16; ++nt) {
      v8f acc = {};
#pragma unroll
      for (int kt = 0; kt < 8; ++kt) {
        v16bf bmat = fb[(kt * 16 + nt) * 32 + l];
        acc = __builtin_amdgcn_wmma_f32_16x16x32_bf16(
            false, pa[kt], false, bmat, (short)0, acc, false, false);
      }
      int n = nt * 16 + m;
#pragma unroll
      for (int j = 0; j < 8; ++j) {
        if (rowE[j] >= 0) {
          float v = acc[j] + TEb[rowTy[j] * Dm + n] + SEb[rowSr[j] * Dm + n];
          out[(size_t)rowE[j] * Dm + n] = fmaxf(v, 0.f);
        }
      }
    }
  }
}

// ---------- launcher ----------
extern "C" void kernel_launch(void* const* d_in, const int* in_sizes, int n_in,
                              void* d_out, int out_size, void* d_ws, size_t ws_size,
                              hipStream_t stream) {
  (void)n_in; (void)out_size; (void)ws_size;
  const int*   type_ids     = (const int*)d_in[0];
  const int*   source_ids   = (const int*)d_in[1];
  const int*   base_ids     = (const int*)d_in[2];
  const float* params       = (const float*)d_in[3];
  const float* type_embed   = (const float*)d_in[4];
  const float* source_embed = (const float*)d_in[5];
  const float* W1 = (const float*)d_in[6];
  const float* b1 = (const float*)d_in[7];
  const float* W2 = (const float*)d_in[8];
  const float* b2 = (const float*)d_in[9];
  const float* Wf = (const float*)d_in[10];
  const float* bf = (const float*)d_in[11];
  float* out = (float*)d_out;
  const int E = in_sizes[0];

  // workspace layout (~1.6 MB total)
  char* ws = (char*)d_ws;
  auto align_up = [](size_t x, size_t a) { return (x + a - 1) & ~(a - 1); };
  int* hist   = (int*)(ws + 0);     // 16 ints
  int* cursor = (int*)(ws + 64);    // 16 ints
  size_t o_order = 256;
  int* order = (int*)(ws + o_order);
  size_t o_w2b = align_up(o_order + (size_t)E * 4, 256);
  __bf16* W2B = (__bf16*)(ws + o_w2b);
  size_t o_wfb = o_w2b + (size_t)(Tt * 8 * 16 * 512) * 2;
  __bf16* WfB = (__bf16*)(ws + o_wfb);
  size_t o_teb = align_up(o_wfb + (size_t)(8 * 16 * 512) * 2, 256);
  float* TEb = (float*)(ws + o_teb);
  float* SEb = (float*)(ws + o_teb + 14 * Dm * 4);

  hipMemsetAsync(ws, 0, 128, stream);  // hist + cursor
  int nb = (E + 255) / 256;
  k_hist   <<<nb, 256, 0, stream>>>(base_ids, E, hist);
  k_scan   <<<1, 32, 0, stream>>>(hist, cursor);
  k_scatter<<<nb, 256, 0, stream>>>(base_ids, E, cursor, order);
  int nprep = Tt * 8 * 16 * 512 + 8 * 16 * 512;
  k_wprep  <<<(nprep + 255) / 256, 256, 0, stream>>>(W2, Wf, W2B, WfB);
  k_embed  <<<(18 * Dm + 255) / 256, 256, 0, stream>>>(type_embed, source_embed, Wf, bf, TEb, SEb);

  int tiles = (E + 15) / 16;
  k_main<<<(tiles + WAVES - 1) / WAVES, WAVES * 32, 0, stream>>>(
      type_ids, source_ids, base_ids, params, W1, b1, b2,
      W2B, WfB, TEb, SEb, order, out, E);
}
